// Attention_60782377173707
// MI455X (gfx1250) — compile-verified, matrix operand order checked by reading, not run on hardware
//
#include <hip/hip_runtime.h>

#define B_   8
#define N_   2048
#define D_   512
#define ROWS (B_ * N_)          // 16384 flat token rows
#define SCALE_ 0.044194173824159216f   // 512^-0.5

typedef __bf16 bf16;
typedef __attribute__((ext_vector_type(16))) __bf16 v16bf;
typedef __attribute__((ext_vector_type(8)))  __bf16 v8bf;
typedef __attribute__((ext_vector_type(4)))  __bf16 v4bf;
typedef __attribute__((ext_vector_type(8)))  float  v8f;
typedef __attribute__((ext_vector_type(4)))  float  v4f;

static __device__ __forceinline__ v8f wmma_bf16(v16bf a, v16bf b, v8f c) {
  // (neg_a, A, neg_b, B, c_mod, C, reuse_a, reuse_b)
  return __builtin_amdgcn_wmma_f32_16x16x32_bf16(false, a, false, b, (short)0, c,
                                                 false, false);
}

// Load one 16-lane-striped bf16 WMMA operand fragment.
// Caller passes p = &row[kbase + half*8]; we grab K{+0..7} and K{+16..23}.
static __device__ __forceinline__ v16bf ldfrag(const bf16* p) {
  union { v16bf v; v8bf h[2]; } u;
  u.h[0] = *(const v8bf*)(p);
  u.h[1] = *(const v8bf*)(p + 16);
  return u.v;
}

// xor-swizzle within 32 lanes (masks <16 stay inside each 16-lane half)
#define SWIZ_F(v, m) \
  __int_as_float(__builtin_amdgcn_ds_swizzle(__float_as_int(v), (((m) << 10) | 0x1f)))

// ---------------------------------------------------------------------------
// fp32 -> bf16 bulk conversion (vectorized by 4)
// ---------------------------------------------------------------------------
__global__ void cvt4_kernel(const float* __restrict__ s, bf16* __restrict__ d, int n4) {
  int i = blockIdx.x * blockDim.x + threadIdx.x;
  int str = gridDim.x * blockDim.x;
  for (; i < n4; i += str) {
    v4f v = *(const v4f*)(s + 4 * (size_t)i);
    v4bf o;
    o[0] = (bf16)v[0]; o[1] = (bf16)v[1]; o[2] = (bf16)v[2]; o[3] = (bf16)v[3];
    *(v4bf*)(d + 4 * (size_t)i) = o;
  }
}

// ---------------------------------------------------------------------------
// QKV projection: qkv = x @ w_qkv^T    (NT GEMM, bf16 WMMA)
// Writes: Qb (pre-scaled), Kb row-major bf16; V transposed bf16 (Vt) + fp32 (Vf)
// grid: (ROWS/16, 1536/128), block 256 (8 waves, one 16-col tile per wave)
// ---------------------------------------------------------------------------
__global__ __launch_bounds__(256) void qkv_kernel(const bf16* __restrict__ xb,
                                                  const bf16* __restrict__ wq,
                                                  bf16* __restrict__ Qb,
                                                  bf16* __restrict__ Kb,
                                                  bf16* __restrict__ Vt,
                                                  float* __restrict__ Vf) {
  const int wave = threadIdx.x >> 5;
  const int lane = threadIdx.x & 31;
  const int l16  = lane & 15;
  const int half = lane >> 4;

  const size_t r0 = (size_t)blockIdx.x * 16;
  const int    j0 = blockIdx.y * 128 + wave * 16;

  const bf16* arow = xb + (r0 + l16) * D_ + half * 8;
  const bf16* brow = wq + (size_t)(j0 + l16) * D_ + half * 8;

  v8f acc = {0.f, 0.f, 0.f, 0.f, 0.f, 0.f, 0.f, 0.f};
#pragma unroll
  for (int c = 0; c < 16; ++c)
    acc = wmma_bf16(ldfrag(arow + c * 32), ldfrag(brow + c * 32), acc);

  const int n = j0 + l16;             // output feature (0..1535), uniform segment per block
  if (n < 512) {                      // Q (pre-scale by SCALE)
#pragma unroll
    for (int i = 0; i < 8; ++i)
      Qb[(r0 + half * 8 + i) * D_ + n] = (bf16)(acc[i] * SCALE_);
  } else if (n < 1024) {              // K
    const int jk = n - 512;
#pragma unroll
    for (int i = 0; i < 8; ++i)
      Kb[(r0 + half * 8 + i) * D_ + jk] = (bf16)acc[i];
  } else {                            // V: fp32 copy + transposed bf16
    const int dv  = n - 1024;
    const int bb  = (int)(r0 / N_);
    const int tk0 = (int)(r0 % N_) + half * 8;
    v8bf pack;
#pragma unroll
    for (int i = 0; i < 8; ++i) {
      float v = acc[i];
      Vf[(r0 + half * 8 + i) * D_ + dv] = v;
      pack[i] = (bf16)v;
    }
    *(v8bf*)(Vt + ((size_t)bb * D_ + dv) * N_ + tk0) = pack;
  }
}

// ---------------------------------------------------------------------------
// Flash attention: one WG per (batch, 16-row tile); 8 waves cover KV blocks of 128.
// Wave w owns S columns [w*16, w*16+16) and O columns [w*64, w*64+64).
// ---------------------------------------------------------------------------
__global__ __launch_bounds__(256) void attn_kernel(const bf16* __restrict__ Qb,
                                                   const bf16* __restrict__ Kb,
                                                   const bf16* __restrict__ Vt,
                                                   bf16* __restrict__ Ob) {
  __shared__ __align__(16) bf16  Pbuf[16 * 128];
  __shared__ float redmax[8][16];
  __shared__ float redsum[8][16];

  const int wg   = blockIdx.x;
  const int b    = wg / (N_ / 16);
  const int t0   = (wg % (N_ / 16)) * 16;
  const size_t rowbase = (size_t)b * N_ + t0;

  const int wave = threadIdx.x >> 5;
  const int lane = threadIdx.x & 31;
  const int l16  = lane & 15;
  const int half = lane >> 4;

  // Q fragments resident in VGPRs for the whole KV sweep (16 x v16bf = 128 VGPRs)
  v16bf qf[16];
  {
    const bf16* qrow = Qb + (rowbase + l16) * D_ + half * 8;
#pragma unroll
    for (int c = 0; c < 16; ++c) qf[c] = ldfrag(qrow + c * 32);
  }

  const bf16* Kbase = Kb + (size_t)b * N_ * D_;
  const bf16* Vbase = Vt + (size_t)b * D_ * (size_t)N_;

  v8f O[4] = {};
  float Mrun[8], Lrun[8];
#pragma unroll
  for (int i = 0; i < 8; ++i) { Mrun[i] = -1e30f; Lrun[i] = 0.f; }

#pragma unroll 1
  for (int it = 0; it < N_ / 128; ++it) {
    const int kv0 = it * 128;

    // ---- S = (Q*scale) @ K^T for this wave's 16 columns ----
    v8f S = {0.f, 0.f, 0.f, 0.f, 0.f, 0.f, 0.f, 0.f};
    {
      const bf16* krow = Kbase + (size_t)(kv0 + wave * 16 + l16) * D_ + half * 8;
#pragma unroll
      for (int c = 0; c < 16; ++c) S = wmma_bf16(qf[c], ldfrag(krow + c * 32), S);
    }

    // ---- per-row max of this 16-col tile (reduce across the 16-lane half) ----
    float tmax[8];
#pragma unroll
    for (int i = 0; i < 8; ++i) {
      float v = S[i];
      v = fmaxf(v, SWIZ_F(v, 1));
      v = fmaxf(v, SWIZ_F(v, 2));
      v = fmaxf(v, SWIZ_F(v, 4));
      v = fmaxf(v, SWIZ_F(v, 8));
      tmax[i] = v;
    }

    __syncthreads();                     // protect LDS reuse from prev iteration
    if (l16 == 0) {
#pragma unroll
      for (int i = 0; i < 8; ++i) redmax[wave][half * 8 + i] = tmax[i];
    }
    __syncthreads();

    // ---- online-softmax row state (replicated identically in all lanes/waves) ----
    float Mnew[8], fcorr[8], p[8];
#pragma unroll
    for (int i = 0; i < 8; ++i) {
      float v = Mrun[i];
#pragma unroll
      for (int w = 0; w < 8; ++w) v = fmaxf(v, redmax[w][half * 8 + i]);
      Mnew[i]  = v;
      fcorr[i] = __expf(Mrun[i] - v);
      Mrun[i]  = v;
    }

    float tsum[8];
#pragma unroll
    for (int i = 0; i < 8; ++i) {
      float pv = __expf(S[i] - Mnew[i]);
      p[i] = pv;
      float s = pv;
      s += SWIZ_F(s, 1);
      s += SWIZ_F(s, 2);
      s += SWIZ_F(s, 4);
      s += SWIZ_F(s, 8);
      tsum[i] = s;
    }
    if (l16 == 0) {
#pragma unroll
      for (int i = 0; i < 8; ++i) redsum[wave][half * 8 + i] = tsum[i];
    }

    // ---- stage P (bf16) into LDS: 16 rows x 128 kv cols ----
#pragma unroll
    for (int i = 0; i < 8; ++i)
      Pbuf[(i + half * 8) * 128 + wave * 16 + l16] = (bf16)p[i];
    __syncthreads();

#pragma unroll
    for (int i = 0; i < 8; ++i) {
      float add = 0.f;
#pragma unroll
      for (int w = 0; w < 8; ++w) add += redsum[w][half * 8 + i];
      Lrun[i] = Lrun[i] * fcorr[i] + add;
    }

    // ---- rescale O, then O += P @ V  (B operand from transposed V, contiguous) ----
#pragma unroll
    for (int dt = 0; dt < 4; ++dt)
#pragma unroll
      for (int i = 0; i < 8; ++i) O[dt][i] *= fcorr[i];

#pragma unroll
    for (int c = 0; c < 4; ++c) {
      v16bf pa = ldfrag(&Pbuf[l16 * 128 + c * 32 + half * 8]);
#pragma unroll
      for (int dt = 0; dt < 4; ++dt) {
        const bf16* vrow =
            Vbase + (size_t)(wave * 64 + dt * 16 + l16) * N_ + kv0 + c * 32 + half * 8;
        O[dt] = wmma_bf16(pa, ldfrag(vrow), O[dt]);
      }
    }
  }

  // ---- normalize and store attention output (bf16, row-major) ----
  float inv[8];
#pragma unroll
  for (int i = 0; i < 8; ++i) inv[i] = 1.0f / Lrun[i];
#pragma unroll
  for (int dt = 0; dt < 4; ++dt) {
    const int d0 = wave * 64 + dt * 16 + l16;
#pragma unroll
    for (int i = 0; i < 8; ++i)
      Ob[(rowbase + half * 8 + i) * D_ + d0] = (bf16)(O[dt][i] * inv[i]);
  }
}

// ---------------------------------------------------------------------------
// Output projection + bias + residual(V):  out = Ob @ w_proj^T + b + Vf
// grid: (ROWS/16, 512/128), block 256
// ---------------------------------------------------------------------------
__global__ __launch_bounds__(256) void proj_kernel(const bf16* __restrict__ Ob,
                                                   const bf16* __restrict__ wp,
                                                   const float* __restrict__ bproj,
                                                   const float* __restrict__ Vf,
                                                   float* __restrict__ out) {
  const int wave = threadIdx.x >> 5;
  const int lane = threadIdx.x & 31;
  const int l16  = lane & 15;
  const int half = lane >> 4;

  const size_t r0 = (size_t)blockIdx.x * 16;
  const int    j0 = blockIdx.y * 128 + wave * 16;

  const bf16* arow = Ob + (r0 + l16) * D_ + half * 8;
  const bf16* brow = wp + (size_t)(j0 + l16) * D_ + half * 8;

  v8f acc = {0.f, 0.f, 0.f, 0.f, 0.f, 0.f, 0.f, 0.f};
#pragma unroll
  for (int c = 0; c < 16; ++c)
    acc = wmma_bf16(ldfrag(arow + c * 32), ldfrag(brow + c * 32), acc);

  const int n = j0 + l16;
  const float bias = bproj[n];
#pragma unroll
  for (int i = 0; i < 8; ++i) {
    const size_t idx = (r0 + half * 8 + i) * D_ + n;
    out[idx] = acc[i] + bias + Vf[idx];
  }
}

// ---------------------------------------------------------------------------
extern "C" void kernel_launch(void* const* d_in, const int* in_sizes, int n_in,
                              void* d_out, int out_size, void* d_ws, size_t ws_size,
                              hipStream_t stream) {
  const float* x      = (const float*)d_in[0];
  const float* w_qkv  = (const float*)d_in[1];
  const float* w_proj = (const float*)d_in[2];
  const float* b_proj = (const float*)d_in[3];
  float*       out    = (float*)d_out;

  char* ws = (char*)d_ws;
  size_t off = 0;
  auto take = [&](size_t bytes) -> char* {
    char* p = ws + off;
    off = (off + bytes + 255) & ~(size_t)255;
    return p;
  };

  bf16*  xb     = (bf16*)take((size_t)ROWS * D_ * 2);
  bf16*  wqkvb  = (bf16*)take((size_t)3 * D_ * D_ * 2);
  bf16*  wprojb = (bf16*)take((size_t)D_ * D_ * 2);
  bf16*  Qb     = (bf16*)take((size_t)ROWS * D_ * 2);
  bf16*  Kb     = (bf16*)take((size_t)ROWS * D_ * 2);
  bf16*  Vt     = (bf16*)take((size_t)ROWS * D_ * 2);
  float* Vf     = (float*)take((size_t)ROWS * D_ * 4);
  bf16*  Ob     = (bf16*)take((size_t)ROWS * D_ * 2);

  cvt4_kernel<<<2048, 256, 0, stream>>>(x, xb, ROWS * D_ / 4);
  cvt4_kernel<<<512, 256, 0, stream>>>(w_qkv, wqkvb, 3 * D_ * D_ / 4);
  cvt4_kernel<<<256, 256, 0, stream>>>(w_proj, wprojb, D_ * D_ / 4);

  qkv_kernel<<<dim3(ROWS / 16, (3 * D_) / 128), 256, 0, stream>>>(xb, wqkvb, Qb, Kb,
                                                                  Vt, Vf);
  attn_kernel<<<dim3(B_ * (N_ / 16)), 256, 0, stream>>>(Qb, Kb, Vt, Ob);
  proj_kernel<<<dim3(ROWS / 16, D_ / 128), 256, 0, stream>>>(Ob, wprojb, b_proj, Vf,
                                                             out);
}
